// MADE_3307124818406
// MI455X (gfx1250) — matrix operand order repaired and checked by reading, not compiled, not run on hardware
//
#include <hip/hip_runtime.h>
#include <hip/hip_bf16.h>

// ---------------------------------------------------------------------------
// MADE autoregressive sampling, persistent-workgroup WMMA implementation.
//   BATCH=2048, INPUT=64, HIDDEN=1024, 64 sequential AR steps.
// Each workgroup owns MT=16 batch rows and runs ALL 64 steps locally (the
// hidden-dim reductions never cross batch rows), so the only synchronization
// is workgroup barriers between layers. Weights are pre-masked + bf16-cast
// once into d_ws; they stay resident in the 192MB L2 for the whole scan.
// Matrix math: v_wmma_f32_16x16x32_bf16 (fp32 accumulate).
// Middle layers run as 2 passes x 4 n-tiles/wave; within a K-step all 4 B
// tiles are loaded into distinct values BEFORE the 4 WMMAs so the loads issue
// as one clause and waits are partial (overlap load latency with WMMA).
// An empty asm memory clobber at the top of each AR step stops LICM from
// hoisting the (const, never-stored) weight loads out of the 64-step loop,
// which previously caused massive scratch spills of B tiles.
// ---------------------------------------------------------------------------

#define INPUT   64
#define HIDDEN  1024
#define BATCH   2048
#define MT      16          // batch rows per workgroup
#define NWAVES  8           // 256 threads = 8 wave32 waves

typedef __attribute__((ext_vector_type(16))) __bf16 bf16x16;
typedef __attribute__((ext_vector_type(8)))  __bf16 bf16x8;
typedef __attribute__((ext_vector_type(8)))  float  f32x8;

// ---- WMMA operand loaders (layouts per CDNA5 ISA 7.12.2, wave32) ----------

// A matrix 16x32 bf16: lane half h, row m=lane%16.
//   v16 elements 0..7  hold K = k0 + h*8      .. +7  (contiguous)
//   v16 elements 8..15 hold K = k0 + 16 + h*8 .. +7  (contiguous)
__device__ __forceinline__ bf16x16 load_a_lds(const __bf16* src, int ldk, int k0) {
    const int lane = threadIdx.x & 31;
    const int m    = lane & 15;
    const int h    = lane >> 4;
    const __bf16* p = src + m * ldk + k0 + h * 8;
    bf16x8 lo = *(const bf16x8*)(p);        // ds_load_b128
    bf16x8 hi = *(const bf16x8*)(p + 16);   // ds_load_b128
    bf16x16 a;
#pragma unroll
    for (int i = 0; i < 8; ++i) { a[i] = lo[i]; a[8 + i] = hi[i]; }
    return a;
}

// B matrix 32x16 bf16 for D = A x B with out[m,n] = sum_k H[m,k]*W[n,k]:
// B[k,n] = W[n,k].  Lane (n = lane%16, h = lane/16) needs K = h*8 + e (e<8)
// and K = 16 + h*8 + (e-8) (e>=8) at fixed n -> two 16B runs of row [n][k].
__device__ __forceinline__ bf16x16 load_b_gl(const __bf16* W,
                                             int ldk, int n0, int k0) {
    const int lane = threadIdx.x & 31;
    const int n    = lane & 15;
    const int h    = lane >> 4;
    const __bf16* p = W + (size_t)(n0 + n) * ldk + k0 + h * 8;
    bf16x8 lo = *(const bf16x8*)(p);        // global_load_b128
    bf16x8 hi = *(const bf16x8*)(p + 16);   // global_load_b128
    bf16x16 b;
#pragma unroll
    for (int i = 0; i < 8; ++i) { b[i] = lo[i]; b[8 + i] = hi[i]; }
    return b;
}

__device__ __forceinline__ f32x8 wmma_bf16(bf16x16 a, bf16x16 b, f32x8 c) {
    return __builtin_amdgcn_wmma_f32_16x16x32_bf16(
        /*neg_a=*/false, a, /*neg_b=*/false, b,
        /*c_mod=*/(short)0, c, /*reuse_a=*/false, /*reuse_b=*/false);
}

// GEMM pass: 4 consecutive 16x16 n-tiles for this wave, full K reduction.
// A from LDS (row-major, leading dim = kdim), B from global weights [n][k].
// All 4 B tiles of a K-step are materialized before the WMMAs: the 8
// global_load_b128 go out as one clause, waits become partial.
__device__ __forceinline__ void gemm4(const __bf16* Asrc,
                                      const __bf16* W,
                                      int kdim, int n0, f32x8 acc[4]) {
#pragma unroll
    for (int t = 0; t < 4; ++t)
#pragma unroll
        for (int r = 0; r < 8; ++r) acc[t][r] = 0.f;
    for (int k0 = 0; k0 < kdim; k0 += 32) {
        bf16x16 a = load_a_lds(Asrc, kdim, k0);
        bf16x16 b[4];
#pragma unroll
        for (int t = 0; t < 4; ++t)
            b[t] = load_b_gl(W, kdim, n0 + t * 16, k0);
        if (k0 + 32 < kdim)   // pull next K-slab of this tile row-block toward L1
            __builtin_prefetch(W + (size_t)n0 * kdim + k0 + 32, 0, 3);
#pragma unroll
        for (int t = 0; t < 4; ++t)
            acc[t] = wmma_bf16(a, b[t], acc[t]);
    }
}

// ---------------------------------------------------------------------------
// Prologue: fold MADE masks into weights, cast to bf16 (row-major [n][k]).
//   d_h[j]  = j % 63 ; m0[j][i] = (d_h[j] >= i)
//   mh[j][k]= (d_h[j] >= d_h[k]) ; mo[o][k] = ((o%64)-1 >= d_h[k])
// ---------------------------------------------------------------------------
__global__ __launch_bounds__(256) void made_prep(
    const float* __restrict__ W0, const float* __restrict__ W1,
    const float* __restrict__ W2, const float* __restrict__ W3,
    __bf16* __restrict__ Wm0, __bf16* __restrict__ Wm1,
    __bf16* __restrict__ Wm2, __bf16* __restrict__ Wm3) {
    const int i = blockIdx.x * 256 + threadIdx.x;

    if (i < HIDDEN * INPUT) {                       // W0: [1024][64]
        int j = i / INPUT, k = i % INPUT;
        float m = ((j % 63) >= k) ? 1.f : 0.f;
        Wm0[i] = (__bf16)(m * W0[i]);
    }
    if (i < HIDDEN * HIDDEN) {                      // W1,W2: [1024][1024]
        int j = i / HIDDEN, k = i % HIDDEN;
        float m = ((j % 63) >= (k % 63)) ? 1.f : 0.f;
        Wm1[i] = (__bf16)(m * W1[i]);
        Wm2[i] = (__bf16)(m * W2[i]);
    }
    if (i < 2 * INPUT * HIDDEN) {                   // W3: [128][1024]
        int o = i / HIDDEN, k = i % HIDDEN;
        float m = (((o % INPUT) - 1) >= (k % 63)) ? 1.f : 0.f;
        Wm3[i] = (__bf16)(m * W3[i]);
    }
}

// ---------------------------------------------------------------------------
// Persistent sampling kernel: one WG = 16 batch rows through all 64 AR steps.
// ---------------------------------------------------------------------------
__global__ __launch_bounds__(256) void made_sample(
    const float* __restrict__ z,
    const float* __restrict__ b0, const float* __restrict__ b1,
    const float* __restrict__ b2, const float* __restrict__ b3,
    const __bf16* Wm0, const __bf16* Wm1,
    const __bf16* Wm2, const __bf16* Wm3,
    float* __restrict__ out) {

    __shared__ __bf16 hbuf[MT][HIDDEN];     // 32 KB  activations (A source)
    __shared__ __bf16 xa[MT][INPUT];        //  2 KB  bf16 current sample x
    __shared__ float  zf[MT][INPUT];        //  4 KB  noise tile
    __shared__ float  th[MT][2 * INPUT];    //  8 KB  theta

    const int tid  = threadIdx.x;
    const int wave = tid >> 5;
    const int lane = tid & 31;
    const int nn   = lane & 15;             // output column within 16x16 tile
    const int hh   = lane >> 4;             // lane half
    const int row0 = blockIdx.x * MT;

    // init x = 0 (bf16 net input), stage z tile
    for (int i = tid; i < MT * INPUT; i += 256) {
        int r = i / INPUT, c = i % INPUT;
        xa[r][c] = (__bf16)0.f;
        zf[r][c] = z[(size_t)(row0 + r) * INPUT + c];
    }

    // biases never change across the 64 steps: preload per-lane into regs
    float bs0[8], bs1[8], bs2[8];
#pragma unroll
    for (int t = 0; t < 8; ++t) {
        int n = (wave * 8 + t) * 16 + nn;
        bs0[t] = b0[n]; bs1[t] = b1[n]; bs2[t] = b2[n];
    }
    const float bs3 = b3[wave * 16 + nn];
    __syncthreads();

    for (int idx = 0; idx < INPUT; ++idx) {
        // Opaque memory clobber: stops LICM from hoisting the loop-invariant
        // weight loads out of the 64-step loop (which spilled B tiles to
        // scratch). Keeps B as in-loop global_load_b128 L2 hits.
        asm volatile("" ::: "memory");

        // ---- layer 0: [16x64] @ Wm0^T -> relu -> hbuf [16x1024] ------
        // hbuf has no concurrent readers here: write each pass immediately.
#pragma unroll
        for (int g = 0; g < 2; ++g) {
            f32x8 acc[4];
            const int n0 = (wave * 8 + g * 4) * 16;
            gemm4(&xa[0][0], Wm0, INPUT, n0, acc);
#pragma unroll
            for (int t = 0; t < 4; ++t) {
#pragma unroll
                for (int r = 0; r < 8; ++r) {
                    float v = acc[t][r] + bs0[g * 4 + t];
                    hbuf[r + 8 * hh][n0 + t * 16 + nn] = (__bf16)(v > 0.f ? v : 0.f);
                }
            }
        }
        __syncthreads();

        // ---- layers 1,2: [16x1024] @ Wm^T -> relu -> hbuf (in place) --
        for (int L = 0; L < 2; ++L) {
            const __bf16* W = L ? Wm2 : Wm1;
            f32x8 acc[2][4];
            for (int g = 0; g < 2; ++g)
                gemm4(&hbuf[0][0], W, HIDDEN, (wave * 8 + g * 4) * 16, acc[g]);
            __syncthreads();           // everyone finished reading hbuf
#pragma unroll
            for (int g = 0; g < 2; ++g) {
#pragma unroll
                for (int t = 0; t < 4; ++t) {
                    int n0 = (wave * 8 + g * 4 + t) * 16;
                    float bias = L ? bs2[g * 4 + t] : bs1[g * 4 + t];
#pragma unroll
                    for (int r = 0; r < 8; ++r) {
                        float v = acc[g][t][r] + bias;
                        hbuf[r + 8 * hh][n0 + nn] = (__bf16)(v > 0.f ? v : 0.f);
                    }
                }
            }
            __syncthreads();
        }

        // ---- layer 3: [16x1024] @ Wm3^T -> theta [16x128] -------------
        // Single n-tile per wave; unroll K by 2 so both k-steps' loads
        // issue ahead of the dependent WMMA chain.
        {
            f32x8 acc;
#pragma unroll
            for (int r = 0; r < 8; ++r) acc[r] = 0.f;
            const int n0 = wave * 16;
            for (int k0 = 0; k0 < HIDDEN; k0 += 64) {
                bf16x16 a0 = load_a_lds(&hbuf[0][0], HIDDEN, k0);
                bf16x16 b0t = load_b_gl(Wm3, HIDDEN, n0, k0);
                bf16x16 a1 = load_a_lds(&hbuf[0][0], HIDDEN, k0 + 32);
                bf16x16 b1t = load_b_gl(Wm3, HIDDEN, n0, k0 + 32);
                acc = wmma_bf16(a0, b0t, acc);
                acc = wmma_bf16(a1, b1t, acc);
            }
#pragma unroll
            for (int r = 0; r < 8; ++r)
                th[r + 8 * hh][n0 + nn] = acc[r] + bs3;
        }
        __syncthreads();

        // ---- AR update: x[:,idx] = z[:,idx]*exp(th_sigma) + th_mu ----
        // Column idx is final: store f32 straight to d_out, update bf16 xa.
        if (tid < MT) {
            float mu = th[tid][idx];
            float sd = __expf(th[tid][INPUT + idx]);
            float xn = zf[tid][idx] * sd + mu;
            out[(size_t)(row0 + tid) * INPUT + idx] = xn;
            xa[tid][idx] = (__bf16)xn;
        }
        __syncthreads();
    }
}

// ---------------------------------------------------------------------------
extern "C" void kernel_launch(void* const* d_in, const int* in_sizes, int n_in,
                              void* d_out, int out_size, void* d_ws, size_t ws_size,
                              hipStream_t stream) {
    const float* z  = (const float*)d_in[0];
    const float* W0 = (const float*)d_in[1];
    const float* b0 = (const float*)d_in[2];
    const float* W1 = (const float*)d_in[3];
    const float* b1 = (const float*)d_in[4];
    const float* W2 = (const float*)d_in[5];
    const float* b2 = (const float*)d_in[6];
    const float* W3 = (const float*)d_in[7];
    const float* b3 = (const float*)d_in[8];

    // workspace layout (bf16): masked weights, row-major [n][k]
    __bf16* Wm0 = (__bf16*)d_ws;                    // 1024 x 64
    __bf16* Wm1 = Wm0 + (size_t)HIDDEN * INPUT;     // 1024 x 1024
    __bf16* Wm2 = Wm1 + (size_t)HIDDEN * HIDDEN;    // 1024 x 1024
    __bf16* Wm3 = Wm2 + (size_t)HIDDEN * HIDDEN;    // 128  x 1024

    made_prep<<<(HIDDEN * HIDDEN + 255) / 256, 256, 0, stream>>>(
        W0, W1, W2, W3, Wm0, Wm1, Wm2, Wm3);

    made_sample<<<BATCH / MT, 256, 0, stream>>>(
        z, b0, b1, b2, b3, Wm0, Wm1, Wm2, Wm3, (float*)d_out);
}
